// Net_27358941675610
// MI455X (gfx1250) — matrix operand order repaired and checked by reading, compiled, not run on hardware
//
#include <hip/hip_runtime.h>

// ---------------------------------------------------------------------------
// MI455X (gfx1250) fused 2-layer MLP via bf16 WMMA.
// The reference's 50-step affine scan collapses analytically (contraction
// factors 0.3 and 0.2 per step -> transients < 1e-24) to:
//   rate = 0.35*sigmoid((6/7)*(x @ W_h^T + b_h))
//   out  = 0.35*sigmoid(0.75*(rate @ W_o^T + b_o) + 0.125*label)
// ---------------------------------------------------------------------------

typedef __bf16 bf16;
typedef __attribute__((ext_vector_type(16))) __bf16 v16bf;
typedef __attribute__((ext_vector_type(8)))  __bf16 v8bf;
typedef __attribute__((ext_vector_type(8)))  float  v8f;

union ABFrag { v16bf v; v8bf h[2]; };

#define IN_SIZE 1024
#define HID     2048
#define OUT_N   512
#define BATCH   4096

// ---------------------------------------------------------------------------
// psp[rows=IN][cols=B] f32  ->  dst[cols=B][rows=IN] bf16  (LDS tile transpose)
// ---------------------------------------------------------------------------
__global__ void __launch_bounds__(256)
k_transpose_cvt(const float* __restrict__ src, bf16* __restrict__ dst,
                int rows, int cols) {
  __shared__ float tile[32][33];
  const int c0 = blockIdx.x * 32;
  const int r0 = blockIdx.y * 32;
  const int tx = threadIdx.x, ty = threadIdx.y;
#pragma unroll
  for (int j = 0; j < 32; j += 8)
    tile[ty + j][tx] = src[(size_t)(r0 + ty + j) * cols + (c0 + tx)];
  __syncthreads();
#pragma unroll
  for (int j = 0; j < 32; j += 8)
    dst[(size_t)(c0 + ty + j) * rows + (r0 + tx)] = (bf16)tile[tx][ty + j];
}

// ---------------------------------------------------------------------------
// plain f32 -> bf16 convert
// ---------------------------------------------------------------------------
__global__ void __launch_bounds__(256)
k_cvt_bf16(const float* __restrict__ s, bf16* __restrict__ d, int n) {
  int i = blockIdx.x * 256 + threadIdx.x;
  if (i < n) d[i] = (bf16)s[i];
}

// ---------------------------------------------------------------------------
// WMMA fragment loads for NT GEMM (A[M][K], B[N][K], both bf16 row-major,
// K-contiguous). Per-lane layouts per CDNA5 ISA 7.12.2:
//   A 16x32: lanes 0-15 hold M=lane, K=0..7 (v0-3) & 16..23 (v4-7);
//            lanes 16-31 hold K=8..15 & 24..31.
//   B 32x16: lanes 0-15 hold N=lane, K=0..7 & 16..23;
//            lanes 16-31 hold K=8..15 & 24..31.
// ---------------------------------------------------------------------------
__device__ __forceinline__ v16bf load_frag_A(const bf16* __restrict__ A, int lda,
                                             int m0, int k0, int lane) {
  const int m    = m0 + (lane & 15);
  const int koff = (lane >> 4) << 3;           // 0 or 8
  const bf16* p  = A + (size_t)m * lda + (k0 + koff);
  ABFrag f;
  f.h[0] = *reinterpret_cast<const v8bf*>(p);        // K = koff .. koff+7
  f.h[1] = *reinterpret_cast<const v8bf*>(p + 16);   // K = koff+16 .. koff+23
  return f.v;
}

__device__ __forceinline__ v16bf load_frag_B(const bf16* __restrict__ B, int ldb,
                                             int n0, int k0, int lane) {
  const int n    = n0 + (lane & 15);
  const int koff = (lane >> 4) << 3;           // 0 or 8
  const bf16* p  = B + (size_t)n * ldb + (k0 + koff);
  ABFrag f;
  f.h[0] = *reinterpret_cast<const v8bf*>(p);        // K = koff .. koff+7
  f.h[1] = *reinterpret_cast<const v8bf*>(p + 16);   // K = koff+16 .. koff+23
  return f.v;
}

// fast sigmoid: v_exp_f32 + v_rcp_f32 (~1 ulp) -- plenty for the tolerance,
// avoids the correctly-rounded v_div_scale/v_fma divide chain.
__device__ __forceinline__ float fast_sigmoid(float x) {
  return __builtin_amdgcn_rcpf(1.0f + __expf(-x));
}

// ---------------------------------------------------------------------------
// C[M][N] = A[M][K] * B[N][K]^T  with fused epilogue.
// One wave computes a 64x64 C tile: 4x4 fragments of 16x16, K-step = 32.
//  -> 16 WMMAs per 8 fragment loads (~32 FLOP/B from L2).
// EPI==1: rate (bf16) = 0.35*sigmoid((6/7)*(acc + bias[n]))
// EPI==2: out  (f32)  = 0.35*sigmoid(0.75*(acc + bias[n]) + 0.125*label[m][n])
// ---------------------------------------------------------------------------
template <int EPI>
__global__ void __launch_bounds__(128)
k_gemm_wmma(const bf16* __restrict__ A, const bf16* __restrict__ Bm,
            const float* __restrict__ bias, const float* __restrict__ label,
            void* __restrict__ Cout, int M, int N, int K) {
  const int lane   = threadIdx.x & 31;
  const int wid    = blockIdx.x * (blockDim.x >> 5) + (threadIdx.x >> 5);
  const int tilesN = N >> 6;                    // N / 64
  const int tm     = (wid / tilesN) * 64;
  const int tn     = (wid % tilesN) * 64;
  if (tm >= M) return;

  v8f c[4][4] = {};                             // 16 accumulator fragments

  for (int k0 = 0; k0 < K; k0 += 32) {
    v16bf a[4], b[4];
#pragma unroll
    for (int mi = 0; mi < 4; ++mi) a[mi] = load_frag_A(A, K, tm + mi * 16, k0, lane);
#pragma unroll
    for (int ni = 0; ni < 4; ++ni) b[ni] = load_frag_B(Bm, K, tn + ni * 16, k0, lane);
#pragma unroll
    for (int mi = 0; mi < 4; ++mi)
#pragma unroll
      for (int ni = 0; ni < 4; ++ni)
        c[mi][ni] = __builtin_amdgcn_wmma_f32_16x16x32_bf16(
            /*neg_a=*/false, a[mi], /*neg_b=*/false, b[ni],
            /*c_mod=*/(short)0, c[mi][ni],
            /*reuse_a=*/false, /*reuse_b=*/false);
  }

  // C/D layout (f32 16x16, 8 VGPRs): vgpr r -> M=r (lanes 0-15), M=8+r (16-31);
  // N = lane & 15.
  const int nl    = lane & 15;
  const int rbase = (lane >> 4) * 8;

#pragma unroll
  for (int mi = 0; mi < 4; ++mi) {
#pragma unroll
    for (int ni = 0; ni < 4; ++ni) {
      const int n    = tn + ni * 16 + nl;
      const float bn = bias[n];
#pragma unroll
      for (int r = 0; r < 8; ++r) {
        const int m   = tm + mi * 16 + rbase + r;
        const float v = c[mi][ni][r] + bn;
        if (EPI == 1) {
          // hidden fixed point: rate = 0.35*sigmoid((6/7)*basal_V_h)
          const float rate = 0.35f * fast_sigmoid(0.857142857142857f * v);
          ((bf16*)Cout)[(size_t)m * N + n] = (bf16)rate;
        } else {
          // output fixed point: soma_o = 0.75*bvo + 0.125*label
          const float soma = 0.75f * v + 0.125f * label[(size_t)m * N + n];
          ((float*)Cout)[(size_t)m * N + n] = 0.35f * fast_sigmoid(soma);
        }
      }
    }
  }
}

// ---------------------------------------------------------------------------
// Inputs (setup_inputs order):
//   0: psp   [IN, B]  f32      3: b_h [HID]      f32
//   1: label [B, OUT] f32      4: W_o [OUT, HID] f32
//   2: W_h   [HID, IN] f32     5: b_o [OUT]      f32
// Output: [B, OUT] f32
// Workspace layout (30 MB):
//   xbf  @ 0        : B*IN   bf16 (8 MB)     wobf @ 12 MB : OUT*HID bf16 (2 MB)
//   whbf @ 8 MB     : HID*IN bf16 (4 MB)     rate @ 14 MB : B*HID   bf16 (16 MB)
// ---------------------------------------------------------------------------
extern "C" void kernel_launch(void* const* d_in, const int* in_sizes, int n_in,
                              void* d_out, int out_size, void* d_ws, size_t ws_size,
                              hipStream_t stream) {
  (void)in_sizes; (void)n_in; (void)out_size; (void)ws_size;
  const float* psp   = (const float*)d_in[0];
  const float* label = (const float*)d_in[1];
  const float* W_h   = (const float*)d_in[2];
  const float* b_h   = (const float*)d_in[3];
  const float* W_o   = (const float*)d_in[4];
  const float* b_o   = (const float*)d_in[5];
  float* out = (float*)d_out;

  char* ws = (char*)d_ws;
  bf16* xbf  = (bf16*)(ws);
  bf16* whbf = (bf16*)(ws + (size_t)8  * 1024 * 1024);
  bf16* wobf = (bf16*)(ws + (size_t)12 * 1024 * 1024);
  bf16* rate = (bf16*)(ws + (size_t)14 * 1024 * 1024);

  // Pack: transpose psp -> xbf[B][IN]; convert weights to bf16.
  k_transpose_cvt<<<dim3(BATCH / 32, IN_SIZE / 32), dim3(32, 8), 0, stream>>>(
      psp, xbf, IN_SIZE, BATCH);
  k_cvt_bf16<<<(HID * IN_SIZE + 255) / 256, 256, 0, stream>>>(W_h, whbf, HID * IN_SIZE);
  k_cvt_bf16<<<(OUT_N * HID + 255) / 256, 256, 0, stream>>>(W_o, wobf, OUT_N * HID);

  // GEMM1: [B,HID] = xbf @ whbf^T, fused hidden-rate epilogue (bf16 out).
  // (4096/64)*(2048/64) = 2048 wave-tiles, 4 waves/block -> 512 blocks.
  k_gemm_wmma<1><<<512, 128, 0, stream>>>(xbf, whbf, b_h, nullptr,
                                          (void*)rate, BATCH, HID, IN_SIZE);

  // GEMM2: [B,OUT] = rate @ wobf^T, fused output epilogue (f32 out).
  // (4096/64)*(512/64) = 512 wave-tiles -> 128 blocks.
  k_gemm_wmma<2><<<128, 128, 0, stream>>>(rate, wobf, b_o, label,
                                          (void*)out, BATCH, OUT_N, HID);
}
